// Conv3dBlock_29171417875121
// MI455X (gfx1250) — matrix operand order repaired and checked by reading, not compiled, not run on hardware
//
#include <hip/hip_runtime.h>
#include <hip/hip_bf16.h>
#include <math.h>

// ---- problem constants (from reference) ----
#define B_  2
#define S_  128
#define T_  128
#define M_  64
#define H_  32
#define W_  32
#define HW_ (H_*W_)
#define G_  8            // groups = 64/8
#define TAPS 27          // 3*3*3
#define GN_EPS 1e-5f

typedef _Float16 f16;
typedef __attribute__((ext_vector_type(16))) _Float16 v16h;
typedef __attribute__((ext_vector_type(8)))  float    v8f;

union V16 { v16h v; uint4 q[2]; };

__device__ __forceinline__ v8f wmma_f16(v16h a, v16h b, v8f c) {
  return __builtin_amdgcn_wmma_f32_16x16x32_f16(false, a, false, b, (short)0, c, false, false);
}

// ---------------------------------------------------------------------------
// 0) zero the f16 grid (uint4 grid-stride)
// ---------------------------------------------------------------------------
__global__ void zero_kernel(uint4* p, int n4) {
  int i = blockIdx.x * blockDim.x + threadIdx.x;
  uint4 z = {0u, 0u, 0u, 0u};
  for (; i < n4; i += gridDim.x * blockDim.x) p[i] = z;
}

// ---------------------------------------------------------------------------
// 1) scatter x (B,S,T,M) f32 -> grid_f16 [b][t][h*32+w][i]
//    one thread per 8-channel chunk of one (b,s,t)
// ---------------------------------------------------------------------------
__global__ void scatter_kernel(const float* __restrict__ x,
                               const int* __restrict__ rows,
                               const int* __restrict__ cols,
                               f16* __restrict__ grid) {
  int idx = blockIdx.x * blockDim.x + threadIdx.x;   // 0 .. B*S*T*8
  if (idx >= B_ * S_ * T_ * 8) return;
  int chunk = idx & 7;
  int r = idx >> 3;
  int t = r % T_;
  int s = (r / T_) % S_;
  int b = r / (T_ * S_);
  const float* src = x + (((size_t)(b * S_ + s) * T_ + t) * M_ + chunk * 8);
  int p = rows[s] * W_ + cols[s];
  f16* dst = grid + (((size_t)(b * T_ + t) * HW_ + p) * M_ + chunk * 8);
  float4 a = *(const float4*)(src);
  float4 c = *(const float4*)(src + 4);
  f16 h[8] = {(f16)a.x,(f16)a.y,(f16)a.z,(f16)a.w,(f16)c.x,(f16)c.y,(f16)c.z,(f16)c.w};
  *(uint4*)dst = *(const uint4*)h;
}

// ---------------------------------------------------------------------------
// 2) conv weights f32 OIDHW -> f16 [o][tap][i]   (tap = dt*9+dh*3+dw)
// ---------------------------------------------------------------------------
__global__ void wprep_kernel(const float* __restrict__ w, f16* __restrict__ wf) {
  int idx = blockIdx.x * blockDim.x + threadIdx.x;   // 64*27*64
  if (idx >= M_ * TAPS * M_) return;
  int i   = idx % M_;
  int tap = (idx / M_) % TAPS;
  int o   = idx / (M_ * TAPS);
  int dt = tap / 9, dh = (tap / 3) % 3, dw = tap % 3;
  wf[idx] = (f16)w[((((size_t)o * M_ + i) * 3 + dt) * 3 + dh) * 3 + dw];
}

__global__ void pwprep_kernel(const float* __restrict__ w, f16* __restrict__ wf) {
  int idx = blockIdx.x * blockDim.x + threadIdx.x;   // 64*64
  if (idx < M_ * M_) wf[idx] = (f16)w[idx];
}

// ---------------------------------------------------------------------------
// 3) conv3d via implicit GEMM with v_wmma_f32_16x16x32_f16
//    block = 128 threads (4 waves); one block per (b, t, output row r)
//    wave og computes out-channels [og*16, og*16+16) for all 32 columns
// ---------------------------------------------------------------------------
#define LPAD 72   // pad inner channel dim 64 -> 72 halfs to spread LDS banks
__global__ void __launch_bounds__(128)
conv_kernel(const f16* __restrict__ grid, const f16* __restrict__ wf,
            const float* __restrict__ conv_b, f16* __restrict__ y) {
  int r = blockIdx.x % H_;
  int t = (blockIdx.x / H_) % T_;
  int b = blockIdx.x / (H_ * T_);

  __shared__ __align__(16) f16 sg[3][3][34][LPAD];   // [dt][rr][cc][i] ~44 KB

  // cooperative fill: 3*3*34 pixels * 8 chunks of 8 halfs = 2448 b128 loads
  for (int q = threadIdx.x; q < 3 * 3 * 34 * 8; q += 128) {
    int chunk = q & 7;
    int pix   = q >> 3;
    int cc = pix % 34;
    int rr = (pix / 34) % 3;
    int dt = pix / (34 * 3);
    int tt = t - 2 + dt;
    int hh = r - 1 + rr;
    int ww = cc - 1;
    uint4 val = {0u, 0u, 0u, 0u};
    if (tt >= 0 && hh >= 0 && hh < H_ && ww >= 0 && ww < W_) {
      val = *(const uint4*)(grid +
            (((size_t)(b * T_ + tt) * HW_ + hh * W_ + ww) * M_ + chunk * 8));
    }
    *(uint4*)(&sg[dt][rr][cc][chunk * 8]) = val;
  }
  __syncthreads();

  const int lane   = threadIdx.x & 31;
  const int og     = threadIdx.x >> 5;
  const int l15    = lane & 15;
  const bool hiH   = lane >= 16;
  const int m      = og * 16 + l15;

  v8f acc0 = {};   // columns 0..15
  v8f acc1 = {};   // columns 16..31

  for (int tap = 0; tap < TAPS; ++tap) {
    int dt = tap / 9, dh = (tap / 3) % 3, dw = tap % 3;
#pragma unroll
    for (int c = 0; c < 2; ++c) {
      // A fragment: 16x32 f16, lane<16 holds K{0..7,16..23}, lane>=16 K{8..15,24..31}
      V16 a;
      const f16* ap = wf + (((size_t)m * TAPS + tap) * M_ + c * 32 + (hiH ? 8 : 0));
      a.q[0] = *(const uint4*)(ap);
      a.q[1] = *(const uint4*)(ap + 16);
      // B fragments: 32x16 f16, lane<16 holds K{0..7,16..23}, lane>=16 K{8..15,24..31}
      int i0 = c * 32 + (hiH ? 8 : 0);
      {
        const f16* bp = &sg[dt][dh][l15 + dw][i0];
        V16 bb; bb.q[0] = *(const uint4*)(bp); bb.q[1] = *(const uint4*)(bp + 16);
        acc0 = wmma_f16(a.v, bb.v, acc0);
      }
      {
        const f16* bp = &sg[dt][dh][16 + l15 + dw][i0];
        V16 bb; bb.q[0] = *(const uint4*)(bp); bb.q[1] = *(const uint4*)(bp + 16);
        acc1 = wmma_f16(a.v, bb.v, acc1);
      }
    }
  }

  // write y[b][t][r*32+col][o] (+bias), 8 consecutive o per lane
  int obase = og * 16 + (hiH ? 8 : 0);
  float4 b0 = *(const float4*)(conv_b + obase);
  float4 b1 = *(const float4*)(conv_b + obase + 4);
  float bias[8] = {b0.x, b0.y, b0.z, b0.w, b1.x, b1.y, b1.z, b1.w};
#pragma unroll
  for (int st = 0; st < 2; ++st) {
    int col = st * 16 + l15;
    v8f acc = st ? acc1 : acc0;
    f16 h[8];
#pragma unroll
    for (int j = 0; j < 8; ++j) h[j] = (f16)(acc[j] + bias[j]);
    *(uint4*)(y + (((size_t)(b * T_ + t) * HW_ + r * W_ + col) * M_ + obase)) =
        *(const uint4*)h;
  }
}

// ---------------------------------------------------------------------------
// 4) GroupNorm stats: per (b,t,g) mean & rstd over 8ch x 1024 positions
//    one block per (b,t); tid -> (o = tid/4, pp = tid%4)
// ---------------------------------------------------------------------------
__global__ void __launch_bounds__(256)
stats_kernel(const f16* __restrict__ y, float* __restrict__ stats) {
  int bt = blockIdx.x;                    // 0 .. B*T-1
  int tid = threadIdx.x;
  int o  = tid >> 2;
  int pp = tid & 3;
  const f16* yp = y + (size_t)bt * HW_ * M_ + o;
  float s1 = 0.f, s2 = 0.f;
  for (int p = pp; p < HW_; p += 4) {
    float v = (float)yp[(size_t)p * M_];
    s1 += v; s2 += v * v;
  }
  __shared__ float r1[256], r2[256];
  r1[tid] = s1; r2[tid] = s2;
  for (int off = 16; off > 0; off >>= 1) {
    __syncthreads();
    if ((tid & 31) < off) { r1[tid] += r1[tid + off]; r2[tid] += r2[tid + off]; }
  }
  __syncthreads();
  if ((tid & 31) == 0) {
    int g = tid >> 5;
    const float n = 8.0f * HW_;
    float mean = r1[tid] / n;
    float var  = r2[tid] / n - mean * mean;
    stats[(size_t)(bt * G_ + g) * 2 + 0] = mean;
    stats[(size_t)(bt * G_ + g) * 2 + 1] = rsqrtf(var + GN_EPS);
  }
}

// ---------------------------------------------------------------------------
// 5) final: gather sensors, GN-normalize, exact GELU, 1x1 conv (WMMA),
//    + pw bias + residual (== x at sensor positions), write (B,S,T,M) f32
//    one block (128 thr = 4 waves) per (b,t)
// ---------------------------------------------------------------------------
__global__ void __launch_bounds__(128)
final_kernel(const f16* __restrict__ y, const f16* __restrict__ pwf,
             const float* __restrict__ pw_b, const float* __restrict__ gamma,
             const float* __restrict__ beta, const float* __restrict__ stats,
             const float* __restrict__ x, const int* __restrict__ rows,
             const int* __restrict__ cols, float* __restrict__ out) {
  int bt = blockIdx.x;
  int t = bt % T_;
  int b = bt / T_;
  int tid = threadIdx.x;

  __shared__ __align__(16) f16 hb[S_][M_];   // gelu(norm(y)) at sensors, 16 KB
  __shared__ float sm[G_], sr[G_];
  if (tid < G_) {
    sm[tid] = stats[(size_t)(bt * G_ + tid) * 2 + 0];
    sr[tid] = stats[(size_t)(bt * G_ + tid) * 2 + 1];
  }
  __syncthreads();

  // phase 1: one thread per sensor, all 64 channels
  {
    int s = tid;                                    // 128 threads = 128 sensors
    int p = rows[s] * W_ + cols[s];
    const f16* yp = y + ((size_t)bt * HW_ + p) * M_;
#pragma unroll 8
    for (int i = 0; i < M_; ++i) {
      float v = (float)yp[i];
      int g = i >> 3;
      float yn = (v - sm[g]) * sr[g] * gamma[i] + beta[i];
      float ge = 0.5f * yn * (1.0f + erff(yn * 0.70710678118654752f));
      hb[s][i] = (f16)ge;
    }
  }
  __syncthreads();

  // phase 2: out[o, s] = pw[o,i] * hb[i, s]  (64x64x128 GEMM via WMMA)
  const int lane = tid & 31;
  const int og   = tid >> 5;
  const int l15  = lane & 15;
  const bool hiH = lane >= 16;
  const int m    = og * 16 + l15;

  V16 a0, a1;                                   // K chunks 0..31 and 32..63
  {
    const f16* ap = pwf + ((size_t)m * M_ + (hiH ? 8 : 0));
    a0.q[0] = *(const uint4*)(ap);      a0.q[1] = *(const uint4*)(ap + 16);
    a1.q[0] = *(const uint4*)(ap + 32); a1.q[1] = *(const uint4*)(ap + 48);
  }
  int obase = og * 16 + (hiH ? 8 : 0);
  float4 p0 = *(const float4*)(pw_b + obase);
  float4 p1 = *(const float4*)(pw_b + obase + 4);
  float pb[8] = {p0.x, p0.y, p0.z, p0.w, p1.x, p1.y, p1.z, p1.w};

  for (int ns = 0; ns < 8; ++ns) {
    int s = ns * 16 + l15;
    int i0 = hiH ? 8 : 0;
    v8f acc = {};
    {
      const f16* bp = &hb[s][i0];
      V16 bb; bb.q[0] = *(const uint4*)(bp); bb.q[1] = *(const uint4*)(bp + 16);
      acc = wmma_f16(a0.v, bb.v, acc);
    }
    {
      const f16* bp = &hb[s][32 + i0];
      V16 bb; bb.q[0] = *(const uint4*)(bp); bb.q[1] = *(const uint4*)(bp + 16);
      acc = wmma_f16(a1.v, bb.v, acc);
    }
    size_t oidx = (((size_t)(b * S_ + s) * T_ + t) * M_ + obase);
    float4 xa = *(const float4*)(x + oidx);
    float4 xb = *(const float4*)(x + oidx + 4);
    float xr[8] = {xa.x, xa.y, xa.z, xa.w, xb.x, xb.y, xb.z, xb.w};
    float4 o0, o1;
    o0.x = acc[0] + pb[0] + xr[0]; o0.y = acc[1] + pb[1] + xr[1];
    o0.z = acc[2] + pb[2] + xr[2]; o0.w = acc[3] + pb[3] + xr[3];
    o1.x = acc[4] + pb[4] + xr[4]; o1.y = acc[5] + pb[5] + xr[5];
    o1.z = acc[6] + pb[6] + xr[6]; o1.w = acc[7] + pb[7] + xr[7];
    *(float4*)(out + oidx)     = o0;
    *(float4*)(out + oidx + 4) = o1;
  }
}

// ---------------------------------------------------------------------------
// launcher
// ---------------------------------------------------------------------------
extern "C" void kernel_launch(void* const* d_in, const int* in_sizes, int n_in,
                              void* d_out, int out_size, void* d_ws, size_t ws_size,
                              hipStream_t stream) {
  const float* x      = (const float*)d_in[0];
  const float* conv_w = (const float*)d_in[1];
  const float* conv_b = (const float*)d_in[2];
  const float* gamma  = (const float*)d_in[3];
  const float* beta   = (const float*)d_in[4];
  const float* pw_w   = (const float*)d_in[5];
  const float* pw_b   = (const float*)d_in[6];
  const int*   rows   = (const int*)d_in[7];
  const int*   cols   = (const int*)d_in[8];
  float* out = (float*)d_out;

  // workspace layout (bytes)
  const size_t GRID_HALFS = (size_t)B_ * T_ * HW_ * M_;        // 16,777,216
  char* ws = (char*)d_ws;
  f16*   grid  = (f16*)(ws);                                    // 33.5 MB
  f16*   y     = (f16*)(ws + GRID_HALFS * 2);                   // 33.5 MB
  f16*   wf    = (f16*)(ws + GRID_HALFS * 4);                   // 221 KB
  f16*   pwf   = (f16*)(ws + GRID_HALFS * 4 + (size_t)M_ * TAPS * M_ * 2);
  float* stats = (float*)(ws + GRID_HALFS * 4 + (size_t)M_ * TAPS * M_ * 2
                             + (size_t)M_ * M_ * 2);

  // 0) zero grid
  {
    int n4 = (int)(GRID_HALFS / 8);
    zero_kernel<<<(n4 + 255) / 256 < 8192 ? (n4 + 255) / 256 : 8192,
                  256, 0, stream>>>((uint4*)grid, n4);
  }
  // 1) scatter x -> grid (f16)
  {
    int n = B_ * S_ * T_ * 8;
    scatter_kernel<<<(n + 255) / 256, 256, 0, stream>>>(x, rows, cols, grid);
  }
  // 2) weight prep
  {
    int n = M_ * TAPS * M_;
    wprep_kernel<<<(n + 255) / 256, 256, 0, stream>>>(conv_w, wf);
    pwprep_kernel<<<(M_ * M_ + 255) / 256, 256, 0, stream>>>(pw_w, pwf);
  }
  // 3) conv3d (WMMA implicit GEMM): one block per (b,t,row)
  conv_kernel<<<B_ * T_ * H_, 128, 0, stream>>>(grid, wf, conv_b, y);
  // 4) GN stats
  stats_kernel<<<B_ * T_, 256, 0, stream>>>(y, stats);
  // 5) fused GN + GELU + pointwise (WMMA) + residual + gather
  final_kernel<<<B_ * T_, 128, 0, stream>>>(y, pwf, pw_b, gamma, beta, stats,
                                            x, rows, cols, out);
}
